// AttentionWithKVCache_43456479101587
// MI455X (gfx1250) — matrix operand order repaired and checked by reading, not compiled, not run on hardware
//
#include <hip/hip_runtime.h>

// ---------------------------------------------------------------------------
// Multi-head attention forward for MI455X (gfx1250), bf16 WMMA + flash attn,
// with Tensor Data Mover (TDM) tile loads into LDS.
// ---------------------------------------------------------------------------

typedef __attribute__((ext_vector_type(16))) __bf16 v16bf;
typedef __attribute__((ext_vector_type(8)))  __bf16 v8bf;
typedef __attribute__((ext_vector_type(8)))  float  v8f;
typedef unsigned int v4u __attribute__((ext_vector_type(4)));
typedef int v8i_t __attribute__((ext_vector_type(8)));
typedef int v4i_t __attribute__((ext_vector_type(4)));

#define D_MODEL 2048
#define N_HEADS 16
#define D_HEAD  128
#define BATCH   4
#define SEQ     2048
#define M_TOTAL (BATCH * SEQ)   // 8192 rows

static __device__ inline v16bf cat8(v8bf lo, v8bf hi) {
    return __builtin_shufflevector(lo, hi, 0, 1, 2, 3, 4, 5, 6, 7,
                                           8, 9, 10, 11, 12, 13, 14, 15);
}

// ---------------------------------------------------------------------------
// TDM: load a 2-D bf16 tile (tile_w elems x tile_h rows, row stride
// row_stride_elems) from global memory into LDS at byte offset lds_addr.
// D# layout per CDNA5 ISA 8.3/8.4 (group0: count/lds/global/type,
// group1: data_size, tensor dims, tile dims, dim0 stride).
// 6-arg builtin form (clang-23 / therock-10.0 headers).
// ---------------------------------------------------------------------------
static __device__ inline void tdm_load_2d(unsigned lds_addr, const void* gptr,
                                          int tile_w_elems, int tile_h,
                                          int row_stride_elems) {
    unsigned long long ga = (unsigned long long)(size_t)gptr;
    v4u g0;
    g0.x = 1u;                                            // count = 1 (valid)
    g0.y = lds_addr;                                      // lds_addr[31:0]
    g0.z = (unsigned)(ga & 0xffffffffull);                // global_addr lo
    g0.w = (unsigned)((ga >> 32) & 0x01ffffffull)         // global_addr hi
           | (2u << 30);                                  // type = 2 (image)
    v8i_t g1;
    g1[0] = 0x00010000;                                   // data_size = 2 B
    g1[1] = (row_stride_elems & 0xffff) << 16;            // tensor_dim0 lo16
    g1[2] = 0xffff << 16;                                 // tensor_dim1 lo16
    g1[3] = (tile_w_elems & 0xffff) << 16;                // tile_dim0
    g1[4] = tile_h & 0xffff;                              // tile_dim1
    g1[5] = row_stride_elems;                             // dim0 stride lo32
    g1[6] = 0;
    g1[7] = 0;
    v4i_t z4 = {0, 0, 0, 0};
    v8i_t z8 = {0, 0, 0, 0, 0, 0, 0, 0};
    __builtin_amdgcn_tensor_load_to_lds(g0, g1, z4, z4, z8, 0);
}

// ---------------------------------------------------------------------------
// fp32 -> bf16 conversion, 8 elements per thread.
// ---------------------------------------------------------------------------
__global__ void cvt_f32_bf16(const float* __restrict__ src,
                             __bf16* __restrict__ dst, int n8) {
    int i = blockIdx.x * blockDim.x + threadIdx.x;
    if (i >= n8) return;
    const float4* s = (const float4*)src;
    float4 a = s[2 * (size_t)i];
    float4 b = s[2 * (size_t)i + 1];
    v8bf o;
    o[0] = (__bf16)a.x; o[1] = (__bf16)a.y; o[2] = (__bf16)a.z; o[3] = (__bf16)a.w;
    o[4] = (__bf16)b.x; o[5] = (__bf16)b.y; o[6] = (__bf16)b.z; o[7] = (__bf16)b.w;
    *(v8bf*)(dst + 8 * (size_t)i) = o;
}

// ---------------------------------------------------------------------------
// C[M,N] = A[M,K] @ W[N,K]^T + bias[N]   (torch Linear), bf16 in, f32 acc.
// Block: 256 threads (8 waves). Block tile 128x128, K-step 32.
// Wave w owns rows [mbase + 16w, +16) x all 128 cols -> 8 wmma accumulators.
// W tile (128 x 32 bf16) loaded into LDS by 8 per-wave TDM descriptors.
// ---------------------------------------------------------------------------
template <bool OUT_F32>
__global__ __launch_bounds__(256) void gemm_wmma(
    const __bf16* __restrict__ A,
    const __bf16* __restrict__ W,
    const float* __restrict__ bias,
    float* __restrict__ outF,
    __bf16* __restrict__ outB,
    int M, int N, int K) {
    __shared__ __align__(16) __bf16 wtile[128 * 32];

    const int tid  = threadIdx.x;
    const int wave = tid >> 5;
    const int lane = tid & 31;
    const int lh   = lane & 15;   // 0..15
    const int kq   = lane >> 4;   // 0 or 1
    const int mbase = blockIdx.y * 128 + wave * 16;
    const int nbase = blockIdx.x * 128;
    const size_t arow = (size_t)(mbase + lh);

    v8f acc[8] = {};

    for (int k0 = 0; k0 < K; k0 += 32) {
        // TDM: this wave DMAs its 16-row slice of the W tile into LDS.
        tdm_load_2d((unsigned)(size_t)(&wtile[wave * 16 * 32]),
                    W + (size_t)(nbase + wave * 16) * K + k0,
                    /*tile_w=*/32, /*tile_h=*/16, /*stride=*/K);

        // A fragment: 16x32, lane lh row, 16 contiguous K at kq*16.
        const __bf16* ap = A + arow * K + k0 + kq * 16;
        v8bf alo = *(const v8bf*)ap;
        v8bf ahi = *(const v8bf*)(ap + 8);
        v16bf afrag = cat8(alo, ahi);

        __builtin_amdgcn_s_wait_tensorcnt(0);
        __syncthreads();

        #pragma unroll
        for (int j = 0; j < 8; ++j) {
            // B fragment: col = j*16 + lh (a row of W), 16 contiguous K per lane.
            v16bf bfrag = *(const v16bf*)(&wtile[(j * 16 + lh) * 32 + kq * 16]);
            acc[j] = __builtin_amdgcn_wmma_f32_16x16x32_bf16(
                false, afrag, false, bfrag, (short)0, acc[j], false, false);
        }
        __syncthreads();
    }

    // Epilogue: bias + store (C layout: VGPR r -> row r + (lane<16?0:8)).
    #pragma unroll
    for (int j = 0; j < 8; ++j) {
        int n = nbase + j * 16 + lh;
        float bv = bias[n];
        #pragma unroll
        for (int r = 0; r < 8; ++r) {
            int m = mbase + r + ((lane < 16) ? 0 : 8);
            float v = acc[j][r] + bv;
            if (OUT_F32) outF[(size_t)m * N + n] = v;
            else         outB[(size_t)m * N + n] = (__bf16)v;
        }
    }
}

// ---------------------------------------------------------------------------
// Flash attention: per (batch, head). Block = 256 threads = 8 waves; each wave
// owns one 16-row Q tile; waves share a 32-key K/V block staged in LDS.
// K tile loaded by TDM (wave 0); V loaded + transposed manually (TDM cannot
// transpose). Online softmax, O accumulated in fp32 wmma frags.
// ---------------------------------------------------------------------------
__global__ __launch_bounds__(256) void flash_attn(
    const __bf16* __restrict__ Q,
    const __bf16* __restrict__ Km,
    const __bf16* __restrict__ Vm,
    __bf16* __restrict__ Ctx) {
    __shared__ __align__(16) __bf16 k_lds[32 * 128];   // [key][d]
    __shared__ __align__(16) __bf16 vT_lds[128 * 32];  // [d][key] (transposed)
    __shared__ __align__(16) __bf16 p_lds[8][16 * 32]; // per-wave P scratch

    const int tid  = threadIdx.x;
    const int wave = tid >> 5;
    const int lane = tid & 31;
    const int lh   = lane & 15;
    const int kq   = lane >> 4;
    const int b    = blockIdx.z;
    const int h    = blockIdx.y;
    const int qtile = blockIdx.x * 8 + wave;
    const size_t rowbase = (size_t)b * SEQ;
    const int colbase = h * D_HEAD;

    // Load Q fragments once: 16x128 = 4 A-frags of 16x32.
    v16bf qfrag[4];
    {
        const int qrow = qtile * 16 + lh;
        const __bf16* qp = Q + (rowbase + qrow) * D_MODEL + colbase;
        #pragma unroll
        for (int d = 0; d < 4; ++d) {
            const __bf16* p = qp + d * 32 + kq * 16;
            v8bf lo = *(const v8bf*)p;
            v8bf hi = *(const v8bf*)(p + 8);
            qfrag[d] = cat8(lo, hi);
        }
    }

    v8f o[8] = {};
    float m_i[8], l_i[8];
    #pragma unroll
    for (int r = 0; r < 8; ++r) { m_i[r] = -3.0e38f; l_i[r] = 0.0f; }

    const float scale = 0.08838834764831845f;  // 1/sqrt(128)

    for (int kb = 0; kb < SEQ; kb += 32) {
        // K tile via TDM (wave 0 issues a single 32x128 descriptor).
        if (wave == 0) {
            tdm_load_2d((unsigned)(size_t)k_lds,
                        Km + (rowbase + kb) * D_MODEL + colbase,
                        /*tile_w=*/128, /*tile_h=*/32, /*stride=*/D_MODEL);
        }
        // V tile: cooperative load + transpose into vT_lds.
        {
            int key  = tid >> 3;
            int doff = (tid & 7) * 16;
            const __bf16* vp = Vm + (rowbase + kb + key) * D_MODEL + colbase + doff;
            v8bf va = *(const v8bf*)vp;
            v8bf vb = *(const v8bf*)(vp + 8);
            #pragma unroll
            for (int e = 0; e < 8; ++e) vT_lds[(doff + e) * 32 + key]     = va[e];
            #pragma unroll
            for (int e = 0; e < 8; ++e) vT_lds[(doff + 8 + e) * 32 + key] = vb[e];
        }
        if (wave == 0) __builtin_amdgcn_s_wait_tensorcnt(0);
        __syncthreads();

        // Scores: S (16x32) = Q (16x128) @ K^T, as two 16x16 wmma accumulators.
        v8f s[2];
        #pragma unroll
        for (int j = 0; j < 2; ++j) {
            s[j] = (v8f){};
            #pragma unroll
            for (int d = 0; d < 4; ++d) {
                v16bf kf = *(const v16bf*)(&k_lds[(j * 16 + lh) * 128 + d * 32 + kq * 16]);
                s[j] = __builtin_amdgcn_wmma_f32_16x16x32_bf16(
                    false, qfrag[d], false, kf, (short)0, s[j], false, false);
            }
        }

        // Online softmax. Row m = r + (lane<16?0:8); N dim lives across 16 lanes,
        // xor masks 1..8 keep the two half-wave groups separate (wave32).
        float sc[8];
        #pragma unroll
        for (int r = 0; r < 8; ++r) {
            float v0 = s[0][r] * scale;
            float v1 = s[1][r] * scale;
            s[0][r] = v0; s[1][r] = v1;
            float mx = fmaxf(v0, v1);
            #pragma unroll
            for (int msk = 1; msk < 16; msk <<= 1)
                mx = fmaxf(mx, __shfl_xor(mx, msk, 32));
            float mnew = fmaxf(m_i[r], mx);
            sc[r] = __expf(m_i[r] - mnew);
            m_i[r] = mnew;
        }
        #pragma unroll
        for (int r = 0; r < 8; ++r) {
            float p0 = __expf(s[0][r] - m_i[r]);
            float p1 = __expf(s[1][r] - m_i[r]);
            s[0][r] = p0; s[1][r] = p1;
            float rs = p0 + p1;
            #pragma unroll
            for (int msk = 1; msk < 16; msk <<= 1)
                rs += __shfl_xor(rs, msk, 32);
            l_i[r] = l_i[r] * sc[r] + rs;
        }

        // C-layout -> A-fragment layout for P via per-wave LDS round trip
        // (same-wave LDS ops are in-order; no barrier needed).
        __bf16* pl = p_lds[wave];
        #pragma unroll
        for (int r = 0; r < 8; ++r) {
            int m = r + ((lane < 16) ? 0 : 8);
            pl[m * 32 + lh]      = (__bf16)s[0][r];
            pl[m * 32 + 16 + lh] = (__bf16)s[1][r];
        }
        v16bf pfrag;
        {
            const __bf16* pp = pl + lh * 32 + kq * 16;
            v8bf lo = *(const v8bf*)pp;
            v8bf hi = *(const v8bf*)(pp + 8);
            pfrag = cat8(lo, hi);
        }

        // Rescale O, then O += P (16x32) @ V (32x128).
        #pragma unroll
        for (int c = 0; c < 8; ++c) {
            #pragma unroll
            for (int r = 0; r < 8; ++r) o[c][r] *= sc[r];
        }
        #pragma unroll
        for (int c = 0; c < 8; ++c) {
            v16bf vf = *(const v16bf*)(&vT_lds[(c * 16 + lh) * 32 + kq * 16]);
            o[c] = __builtin_amdgcn_wmma_f32_16x16x32_bf16(
                false, pfrag, false, vf, (short)0, o[c], false, false);
        }
        __syncthreads();
    }

    // Epilogue: normalize by l_i and store ctx (bf16) back in [b*S+s, h*128+d].
    #pragma unroll
    for (int c = 0; c < 8; ++c) {
        #pragma unroll
        for (int r = 0; r < 8; ++r) {
            float v = o[c][r] / l_i[r];
            int m = qtile * 16 + r + ((lane < 16) ? 0 : 8);
            int d = c * 16 + lh;
            Ctx[(rowbase + m) * D_MODEL + colbase + d] = (__bf16)v;
        }
    }
}

// ---------------------------------------------------------------------------
// Host entry point.
// ---------------------------------------------------------------------------
extern "C" void kernel_launch(void* const* d_in, const int* in_sizes, int n_in,
                              void* d_out, int out_size, void* d_ws, size_t ws_size,
                              hipStream_t stream) {
    const float* x    = (const float*)d_in[0];
    const float* wq_w = (const float*)d_in[1];
    const float* wq_b = (const float*)d_in[2];
    const float* wk_w = (const float*)d_in[3];
    const float* wk_b = (const float*)d_in[4];
    const float* wv_w = (const float*)d_in[5];
    const float* wv_b = (const float*)d_in[6];
    const float* wo_w = (const float*)d_in[7];
    const float* wo_b = (const float*)d_in[8];

    const size_t xElems = (size_t)M_TOTAL * D_MODEL;   // 16.7M
    const size_t wElems = (size_t)D_MODEL * D_MODEL;   // 4.2M

    char* ws = (char*)d_ws;
    __bf16* xb  = (__bf16*)ws; ws += xElems * 2;
    __bf16* wqb = (__bf16*)ws; ws += wElems * 2;
    __bf16* wkb = (__bf16*)ws; ws += wElems * 2;
    __bf16* wvb = (__bf16*)ws; ws += wElems * 2;
    __bf16* wob = (__bf16*)ws; ws += wElems * 2;
    __bf16* Qb  = (__bf16*)ws; ws += xElems * 2;
    __bf16* Kb  = (__bf16*)ws; ws += xElems * 2;
    __bf16* Vb  = (__bf16*)ws; ws += xElems * 2;
    __bf16* Cb  = (__bf16*)ws; ws += xElems * 2;

    // 1) fp32 -> bf16 conversions.
    {
        int n8 = (int)(xElems / 8);
        cvt_f32_bf16<<<(n8 + 255) / 256, 256, 0, stream>>>(x, xb, n8);
        int w8 = (int)(wElems / 8);
        cvt_f32_bf16<<<(w8 + 255) / 256, 256, 0, stream>>>(wq_w, wqb, w8);
        cvt_f32_bf16<<<(w8 + 255) / 256, 256, 0, stream>>>(wk_w, wkb, w8);
        cvt_f32_bf16<<<(w8 + 255) / 256, 256, 0, stream>>>(wv_w, wvb, w8);
        cvt_f32_bf16<<<(w8 + 255) / 256, 256, 0, stream>>>(wo_w, wob, w8);
    }

    // 2) Q/K/V projections (bf16 out).
    dim3 gg(D_MODEL / 128, M_TOTAL / 128);  // (16, 64)
    gemm_wmma<false><<<gg, 256, 0, stream>>>(xb, wqb, wq_b, (float*)nullptr, Qb,
                                             M_TOTAL, D_MODEL, D_MODEL);
    gemm_wmma<false><<<gg, 256, 0, stream>>>(xb, wkb, wk_b, (float*)nullptr, Kb,
                                             M_TOTAL, D_MODEL, D_MODEL);
    gemm_wmma<false><<<gg, 256, 0, stream>>>(xb, wvb, wv_b, (float*)nullptr, Vb,
                                             M_TOTAL, D_MODEL, D_MODEL);

    // 3) Flash attention: grid (q-tile groups, heads, batch).
    flash_attn<<<dim3(SEQ / (16 * 8), N_HEADS, BATCH), 256, 0, stream>>>(
        Qb, Kb, Vb, Cb);

    // 4) Output projection (fp32 out -> d_out).
    gemm_wmma<true><<<gg, 256, 0, stream>>>(Cb, wob, wo_b, (float*)d_out,
                                            (__bf16*)nullptr,
                                            M_TOTAL, D_MODEL, D_MODEL);
}
